// MVArcFace_54898271977563
// MI455X (gfx1250) — matrix operand order repaired and checked by reading, not compiled, not run on hardware
//
#include <hip/hip_runtime.h>
#include <hip/hip_bf16.h>
#include <math.h>

typedef __attribute__((ext_vector_type(16))) _Float16 v16h;
typedef __attribute__((ext_vector_type(8)))  float    v8f;

#define B_SZ 512
#define D_SZ 512
#define C_SZ 100000
#define TC 128          // classes per workgroup
#define TB 64           // batch cols per workgroup
#define KC 32           // K chunk (one f16 WMMA K)
#define NT ((C_SZ + TC - 1) / TC)   // 782 class tiles

#define SCALE_F 64.0f
#define COS_M_F 0.8775825618903728f
#define SIN_M_F 0.4794255386042030f
#define TH_F   (-0.8775825618903728f)
#define MM_F    0.2397127693021015f
#define T_F     0.2f
#define EPS_F   1e-7f
#define NEG_BIG (-1.0e30f)

// ---- optional CDNA5 async global->LDS path (guarded; falls back to sync copy)
#if defined(__has_builtin)
#  if __has_builtin(__builtin_amdgcn_global_load_async_to_lds_b128)
#    define HAVE_ASYNC_LDS 1
#  endif
#endif

#ifdef HAVE_ASYNC_LDS
// Signature (from hipcc diagnostic): param0 = addrspace(1) v4i* (prints as
// "__vector_size__(4*sizeof(int)) int __device__ *"), param1 = LDS side,
// params 2/3 = imm offset / cpol.
typedef int v4i_b __attribute__((vector_size(16)));
typedef __attribute__((address_space(1))) v4i_b* as1_v4i;
typedef __attribute__((address_space(3))) v4i_b* as3_v4i;

__device__ __forceinline__ void async_cpy_b128(const void* gp, void* lp) {
  __builtin_amdgcn_global_load_async_to_lds_b128((as1_v4i)gp, (as3_v4i)lp, 0, 0);
}
__device__ __forceinline__ void async_wait0() {
#if __has_builtin(__builtin_amdgcn_s_wait_asynccnt)
  __builtin_amdgcn_s_wait_asynccnt(0);
#else
  asm volatile("s_wait_asynccnt 0" ::: "memory");
#endif
}
#endif

// ---------------- Kernel 1: normalize embeddings, split to f16 hi/lo ----------
__global__ void k_prep_e(const float* __restrict__ emb,
                         _Float16* __restrict__ eh, _Float16* __restrict__ el) {
  int b = blockIdx.x, t = threadIdx.x;             // 256 threads, 2 elems each
  const float* row = emb + (size_t)b * D_SZ;
  float v0 = row[t], v1 = row[t + 256];
  __shared__ float red[256];
  __shared__ float invn;
  red[t] = v0 * v0 + v1 * v1;
  __syncthreads();
  for (int s = 128; s > 0; s >>= 1) { if (t < s) red[t] += red[t + s]; __syncthreads(); }
  if (t == 0) invn = 1.0f / sqrtf(red[0]);
  __syncthreads();
  float inv = invn;
  float x0 = v0 * inv, x1 = v1 * inv;
  _Float16 h0 = (_Float16)x0, h1 = (_Float16)x1;
  size_t base = (size_t)b * D_SZ;
  eh[base + t]       = h0;
  eh[base + t + 256] = h1;
  el[base + t]       = (_Float16)(x0 - (float)h0);
  el[base + t + 256] = (_Float16)(x1 - (float)h1);
}

// ---------------- Kernel 2: 1/||w_c|| for every class -------------------------
__global__ void k_wnorm(const float* __restrict__ W, float* __restrict__ wninv) {
  int c = blockIdx.x, t = threadIdx.x;             // 128 threads, float4 each
  const float4* row = (const float4*)(W + (size_t)c * D_SZ);
  float4 f = row[t];
  __shared__ float red[128];
  red[t] = f.x * f.x + f.y * f.y + f.z * f.z + f.w * f.w;
  __syncthreads();
  for (int s = 64; s > 0; s >>= 1) { if (t < s) red[t] += red[t + s]; __syncthreads(); }
  if (t == 0) wninv[c] = 1.0f / sqrtf(red[0]);
}

// ---------------- Kernel 3: pos[b] = <e_b, w_gt>, phi[b] ----------------------
__global__ void k_posphi(const float* __restrict__ W,
                         const _Float16* __restrict__ eh, const _Float16* __restrict__ el,
                         const float* __restrict__ wninv, const int* __restrict__ gt,
                         float* __restrict__ phi) {
  int b = blockIdx.x, t = threadIdx.x;             // 128 threads
  int g = gt[b];
  float winv = wninv[g];
  float acc = 0.0f;
  for (int d = t; d < D_SZ; d += 128) {
    float e = (float)eh[(size_t)b * D_SZ + d] + (float)el[(size_t)b * D_SZ + d];
    acc += e * (W[(size_t)g * D_SZ + d] * winv);
  }
  __shared__ float red[128];
  red[t] = acc; __syncthreads();
  for (int s = 64; s > 0; s >>= 1) { if (t < s) red[t] += red[t + s]; __syncthreads(); }
  if (t == 0) {
    float pos = fminf(fmaxf(red[0], -1.0f + EPS_F), 1.0f - EPS_F);
    float st  = sqrtf(fminf(fmaxf(1.0f - pos * pos, EPS_F), 1.0f - EPS_F));
    float ph  = pos * COS_M_F - st * SIN_M_F;
    phi[b] = (pos > TH_F) ? ph : (pos - MM_F);
  }
}

// ---------------- Kernel 4: WMMA GEMM + margin + per-tile logsumexp ----------
// 4 waves (128 threads): 2x2 wave grid, each wave computes 64 classes x 32 batch.
__global__ __launch_bounds__(128) void k_gemm(
    const float* __restrict__ W,
    const _Float16* __restrict__ eh, const _Float16* __restrict__ el,
    const float* __restrict__ wninv, const float* __restrict__ phi,
    const int* __restrict__ gt,
    float* __restrict__ pmax, float* __restrict__ psum) {
  __shared__ _Float16 sWh[TC * KC], sWl[TC * KC];   // normalized W tile, hi/lo f16
  __shared__ _Float16 sEh[TB * KC], sEl[TB * KC];   // normalized E tile, hi/lo f16
  __shared__ float sWn[TC];
  __shared__ float cm[2 * TB], cs[2 * TB];

  const int tid  = threadIdx.x;
  const int lane = tid & 31;
  const int wave = tid >> 5;
  const int wM = wave >> 1, wN = wave & 1;
  const int c0 = blockIdx.x * TC;
  const int b0 = blockIdx.y * TB;

  if (tid < TC) sWn[tid] = (c0 + tid < C_SZ) ? wninv[c0 + tid] : 0.0f;
  __syncthreads();

  v8f acc[4][2];
  #pragma unroll
  for (int mi = 0; mi < 4; ++mi)
    #pragma unroll
    for (int ni = 0; ni < 2; ++ni)
      #pragma unroll
      for (int r = 0; r < 8; ++r) acc[mi][ni][r] = 0.0f;

  for (int kk = 0; kk < D_SZ; kk += KC) {
#ifdef HAVE_ASYNC_LDS
    // E chunk: pure copy -> async DMA to LDS (ASYNCcnt), overlapped with the
    // VALU-bound W transform below. Each thread: row=tid/2, 16 halves = 2xB128.
    {
      int r  = tid >> 1;
      int kf = (tid & 1) << 4;
      size_t gbase = (size_t)(b0 + r) * D_SZ + kk + kf;
      int    lbase = r * KC + kf;
      async_cpy_b128(&eh[gbase], &sEh[lbase]);
      async_cpy_b128(&eh[gbase + 8], &sEh[lbase + 8]);
      async_cpy_b128(&el[gbase], &sEl[lbase]);
      async_cpy_b128(&el[gbase + 8], &sEl[lbase + 8]);
    }
#endif
    // Cooperative stage: W chunk (normalize + f32->f16 hi/lo split) — VALU path
    #pragma unroll 4
    for (int i = tid; i < TC * KC; i += 128) {
      int r = i >> 5, k = i & 31;
      int cg = c0 + r;
      float v = 0.0f;
      if (cg < C_SZ) v = W[(size_t)cg * D_SZ + kk + k] * sWn[r];
      _Float16 h = (_Float16)v;
      sWh[i] = h;
      sWl[i] = (_Float16)(v - (float)h);
    }
#ifndef HAVE_ASYNC_LDS
    // Fallback: synchronous E chunk copy through VGPRs
    #pragma unroll 2
    for (int i = tid; i < TB * KC; i += 128) {
      int r = i >> 5, k = i & 31;
      size_t gi = (size_t)(b0 + r) * D_SZ + kk + k;
      sEh[i] = eh[gi];
      sEl[i] = el[gi];
    }
#else
    async_wait0();                          // s_wait_asynccnt 0
#endif
    __syncthreads();

    if (kk + KC < D_SZ) {
      int pc = c0 + tid; if (pc >= C_SZ) pc = C_SZ - 1;
      __builtin_prefetch(&W[(size_t)pc * D_SZ + kk + KC], 0, 1);  // global_prefetch_b8
    }

    // B fragments (batch): 32x16 f16 B layout — lane = column, halves = K
    v16h bh[2], bl[2];
    #pragma unroll
    for (int ni = 0; ni < 2; ++ni) {
      int n  = wN * 32 + ni * 16 + (lane & 15);
      int kb = (lane >> 4) << 4;                 // lanes 16-31 hold K=16..31
      #pragma unroll
      for (int h = 0; h < 16; ++h) {
        bh[ni][h] = sEh[n * KC + kb + h];
        bl[ni][h] = sEl[n * KC + kb + h];
      }
    }
    #pragma unroll
    for (int mi = 0; mi < 4; ++mi) {
      // A fragment (classes): 16x32 f16 A layout, contiguous K (matches B)
      int m  = wM * 64 + mi * 16 + (lane & 15);
      int kb = (lane >> 4) << 4;                 // lanes 16-31 hold K=16..31
      v16h ah, al;
      #pragma unroll
      for (int h = 0; h < 16; ++h) {
        int k = kb + h;
        ah[h] = sWh[m * KC + k];
        al[h] = sWl[m * KC + k];
      }
      #pragma unroll
      for (int ni = 0; ni < 2; ++ni) {
        // split-precision: hi*hi + hi*lo + lo*hi, all f32-accumulated
        acc[mi][ni] = __builtin_amdgcn_wmma_f32_16x16x32_f16(false, ah, false, bh[ni], (short)0, acc[mi][ni], false, false);
        acc[mi][ni] = __builtin_amdgcn_wmma_f32_16x16x32_f16(false, ah, false, bl[ni], (short)0, acc[mi][ni], false, false);
        acc[mi][ni] = __builtin_amdgcn_wmma_f32_16x16x32_f16(false, al, false, bh[ni], (short)0, acc[mi][ni], false, false);
      }
    }
    __syncthreads();
  }

  // ---------------- Margin epilogue + per-tile logsumexp partials ------------
  #pragma unroll
  for (int ni = 0; ni < 2; ++ni) {
    int bcol = b0 + wN * 32 + ni * 16 + (lane & 15);
    float ph = phi[bcol];
    int   g  = gt[bcol];
    float m = NEG_BIG;
    #pragma unroll
    for (int mi = 0; mi < 4; ++mi)
      #pragma unroll
      for (int r = 0; r < 8; ++r) {
        // C/D layout: VGPR r -> M=r (lanes 0-15) / M=r+8 (lanes 16-31), N=lane&15
        int cg = c0 + wM * 64 + mi * 16 + r + ((lane >= 16) ? 8 : 0);
        float v = acc[mi][ni][r];
        v = fminf(fmaxf(v, -1.0f + EPS_F), 1.0f - EPS_F);
        float lg;
        if (cg >= C_SZ)      lg = NEG_BIG;
        else if (cg == g)    lg = SCALE_F * ph;                          // scatter
        else                 lg = SCALE_F * ((v > ph) ? ((T_F + 1.0f) * v + T_F) : v);
        acc[mi][ni][r] = lg;
        m = fmaxf(m, lg);
      }
    float s = 0.0f;
    #pragma unroll
    for (int mi = 0; mi < 4; ++mi)
      #pragma unroll
      for (int r = 0; r < 8; ++r) {
        float lg = acc[mi][ni][r];
        s += (lg <= 0.5f * NEG_BIG) ? 0.0f : expf(lg - m);
      }
    // lanes l and l^16 hold the SAME batch column, different classes: combine
    float m2 = __shfl_xor(m, 16, 32);
    float s2 = __shfl_xor(s, 16, 32);
    float M = fmaxf(m, m2);
    float S = s * expf(m - M) + s2 * expf(m2 - M);   // s==0 guards dead halves
    if (lane < 16) {
      int bl_ = wN * 32 + ni * 16 + lane;
      cm[wM * TB + bl_] = M;
      cs[wM * TB + bl_] = S;
    }
  }
  __syncthreads();
  // combine across the two class-wave rows, emit per-tile (max, sumexp)
  if (tid < TB) {
    float m0 = cm[tid], m1 = cm[TB + tid];
    float s0 = cs[tid], s1 = cs[TB + tid];
    float M = fmaxf(m0, m1);
    float S = s0 * expf(m0 - M) + s1 * expf(m1 - M);
    pmax[(size_t)blockIdx.x * B_SZ + b0 + tid] = M;
    psum[(size_t)blockIdx.x * B_SZ + b0 + tid] = S;
  }
}

// ---------------- Kernel 5: streaming LSE over the 782 tiles per row ---------
__global__ void k_rowlse(const float* __restrict__ pmax, const float* __restrict__ psum,
                         const float* __restrict__ phi, float* __restrict__ perb) {
  int b = blockIdx.x, t = threadIdx.x;             // 256 threads
  __shared__ float red[256];
  __shared__ float Msh;
  float m = NEG_BIG;
  for (int i = t; i < NT; i += 256) m = fmaxf(m, pmax[(size_t)i * B_SZ + b]);
  red[t] = m; __syncthreads();
  for (int s = 128; s > 0; s >>= 1) { if (t < s) red[t] = fmaxf(red[t], red[t + s]); __syncthreads(); }
  if (t == 0) Msh = red[0];
  __syncthreads();
  float M = Msh;
  float s = 0.0f;
  for (int i = t; i < NT; i += 256) {
    float mi = pmax[(size_t)i * B_SZ + b];
    float si = psum[(size_t)i * B_SZ + b];
    s += si * expf(mi - M);
  }
  __syncthreads();
  red[t] = s; __syncthreads();
  for (int st = 128; st > 0; st >>= 1) { if (t < st) red[t] += red[t + st]; __syncthreads(); }
  if (t == 0) perb[b] = (M + logf(red[0])) - SCALE_F * phi[b];   // -log p_gt
}

// ---------------- Kernel 6: mean over the batch ------------------------------
__global__ void k_final(const float* __restrict__ perb, float* __restrict__ out) {
  int t = threadIdx.x;                             // 512 threads
  __shared__ float red[512];
  red[t] = perb[t];
  __syncthreads();
  for (int s = 256; s > 0; s >>= 1) { if (t < s) red[t] += red[t + s]; __syncthreads(); }
  if (t == 0) out[0] = red[0] / (float)B_SZ;
}

extern "C" void kernel_launch(void* const* d_in, const int* in_sizes, int n_in,
                              void* d_out, int out_size, void* d_ws, size_t ws_size,
                              hipStream_t stream) {
  const float* emb = (const float*)d_in[0];
  const float* W   = (const float*)d_in[1];
  const int*   gt  = (const int*)d_in[2];
  float* out = (float*)d_out;

  // workspace layout (~4.6 MB total)
  _Float16* EH = (_Float16*)d_ws;
  _Float16* EL = EH + (size_t)B_SZ * D_SZ;
  float* F     = (float*)((char*)d_ws + 2u * sizeof(_Float16) * (size_t)B_SZ * D_SZ);
  float* wninv = F;                                 // 100000
  float* phi   = wninv + 100352;                    // 512
  float* perb  = phi + 512;                         // 512
  float* pmax  = perb + 512;                        // NT * 512
  float* psum  = pmax + (size_t)NT * B_SZ;          // NT * 512

  k_prep_e<<<B_SZ, 256, 0, stream>>>(emb, EH, EL);
  k_wnorm <<<C_SZ, 128, 0, stream>>>(W, wninv);
  k_posphi<<<B_SZ, 128, 0, stream>>>(W, EH, EL, wninv, gt, phi);
  dim3 g(NT, B_SZ / TB);
  k_gemm  <<<g, 128, 0, stream>>>(W, EH, EL, wninv, phi, gt, pmax, psum);
  k_rowlse<<<B_SZ, 256, 0, stream>>>(pmax, psum, phi, perb);
  k_final <<<1, B_SZ, 0, stream>>>(perb, out);
}
